// QLlamaDecoderLayer_72275709657122
// MI455X (gfx1250) — compile-verified, matrix operand order checked by reading, not run on hardware
//
#include <hip/hip_runtime.h>
#include <hip/hip_bf16.h>
#include <cstdint>

// ---------------------------------------------------------------------------
// Types / helpers
// ---------------------------------------------------------------------------
typedef __bf16 bf16_t;
typedef __attribute__((ext_vector_type(16))) __bf16 bf16x16;
typedef __attribute__((ext_vector_type(8)))  __bf16 bf16x8;
typedef __attribute__((ext_vector_type(8)))  float  f32x8;
typedef int gvec4i __attribute__((vector_size(16)));   // matches builtin proto

#define AS1 __attribute__((address_space(1)))
#define AS3 __attribute__((address_space(3)))

#if defined(__has_builtin)
#if __has_builtin(__builtin_amdgcn_global_load_async_to_lds_b128) && \
    __has_builtin(__builtin_amdgcn_s_wait_asynccnt)
#define USE_ASYNC 1
#endif
#endif
#ifndef USE_ASYNC
#define USE_ASYNC 0
#endif

__device__ __forceinline__ bf16_t f2bf(float f) {
  unsigned u = __builtin_bit_cast(unsigned, f);
  unsigned r = (u + 0x7FFFu + ((u >> 16) & 1u)) >> 16;   // RNE
  unsigned short s = (unsigned short)r;
  return __builtin_bit_cast(bf16_t, s);
}
__device__ __forceinline__ float bf2f(bf16_t h) {
  unsigned short s = __builtin_bit_cast(unsigned short, h);
  unsigned u = ((unsigned)s) << 16;
  return __builtin_bit_cast(float, u);
}

// 16B global -> LDS copy (async path on CDNA5, sync fallback otherwise)
__device__ __forceinline__ void copy16_g2l(bf16_t* l, const bf16_t* g) {
#if USE_ASYNC
  __builtin_amdgcn_global_load_async_to_lds_b128(
      (AS1 gvec4i*)(uintptr_t)g,
      (AS3 gvec4i*)(uint32_t)(uintptr_t)l, 0, 0);
#else
  *(bf16x8*)l = *(const bf16x8*)g;
#endif
}

// Load a 16x32 bf16 A/B fragment row (per-lane): CDNA5 layout — lanes 0-15
// hold K {0..7,16..23}, lanes 16-31 hold K {8..15,24..31} of their row.
__device__ __forceinline__ bf16x16 load_frag(const bf16_t* base, int lh) {
  union { bf16x16 v; bf16x8 h[2]; } u;
  u.h[0] = *(const bf16x8*)(base + lh * 8);
  u.h[1] = *(const bf16x8*)(base + 16 + lh * 8);
  return u.v;
}

__device__ __forceinline__ f32x8 wmma_bf16(bf16x16 a, bf16x16 b, f32x8 c) {
  return __builtin_amdgcn_wmma_f32_16x16x32_bf16(false, a, false, b,
                                                 (short)0, c, false, false);
}

// ---------------------------------------------------------------------------
// RMSNorm (one row of 2048 per block)
// ---------------------------------------------------------------------------
__global__ __launch_bounds__(256) void rmsnorm_k(const float* __restrict__ X,
                                                 const float* __restrict__ W,
                                                 float* __restrict__ Y) {
  constexpr int Hd = 2048;
  size_t row = blockIdx.x;
  const float* x = X + row * Hd;
  float ss = 0.f;
  for (int i = threadIdx.x; i < Hd; i += 256) { float v = x[i]; ss += v * v; }
  for (int off = 16; off; off >>= 1) ss += __shfl_xor(ss, off, 32);
  __shared__ float red[8];
  int wid = threadIdx.x >> 5, lane = threadIdx.x & 31;
  if (!lane) red[wid] = ss;
  __syncthreads();
  float tot = red[0];
#pragma unroll
  for (int i = 1; i < 8; ++i) tot += red[i];
  float inv = rsqrtf(tot * (1.f / Hd) + 1e-5f);
  float* y = Y + row * Hd;
  for (int i = threadIdx.x; i < Hd; i += 256) y[i] = x[i] * inv * W[i];
}

// ---------------------------------------------------------------------------
// Gather (column permutation) + per-128-group asymmetric 4-bit fake quant,
// emit bf16 in permuted order.  quant_all=1 -> weights; 0 -> skip last group.
// grid = (rows, K/128), block = 128.
// ---------------------------------------------------------------------------
__global__ __launch_bounds__(128) void reorder_quant_k(
    const float* __restrict__ X, const int* __restrict__ idx,
    bf16_t* __restrict__ Xq, int K, int quant_all) {
  size_t row = blockIdx.x;
  int g = blockIdx.y, t = threadIdx.x;
  int colp = (g << 7) + t;
  int src = idx[colp];
  float v = X[row * (size_t)K + src];
  float mx = v, mn = v;
  int lane = t & 31, wid = t >> 5;
  for (int off = 16; off; off >>= 1) {
    mx = fmaxf(mx, __shfl_xor(mx, off, 32));
    mn = fminf(mn, __shfl_xor(mn, off, 32));
  }
  __shared__ float smx[4], smn[4];
  if (!lane) { smx[wid] = mx; smn[wid] = mn; }
  __syncthreads();
  mx = fmaxf(fmaxf(smx[0], smx[1]), fmaxf(smx[2], smx[3]));
  mn = fminf(fminf(smn[0], smn[1]), fminf(smn[2], smn[3]));
  float out = v;
  if (quant_all || (g + 1 < (int)gridDim.y)) {
    float scale = fmaxf(mx - mn, 1e-5f) * (1.f / 15.f);
    float inv = 1.f / scale;
    float base = fminf(fmaxf(rintf(-mn * inv), 0.f), 15.f);
    out = (fminf(fmaxf(rintf(v * inv) + base, 0.f), 15.f) - base) * scale;
  }
  Xq[row * (size_t)K + colp] = f2bf(out);
}

// ---------------------------------------------------------------------------
// Fused SiLU(g)*u  -> gather by idx -> group fake-quant -> bf16 (skip last grp)
// ---------------------------------------------------------------------------
__global__ __launch_bounds__(128) void silu_mul_quant_k(
    const bf16_t* __restrict__ G, const bf16_t* __restrict__ U,
    const int* __restrict__ idx, bf16_t* __restrict__ Dq, int K) {
  size_t row = blockIdx.x;
  int g = blockIdx.y, t = threadIdx.x;
  int colp = (g << 7) + t;
  int src = idx[colp];
  float gv = bf2f(G[row * (size_t)K + src]);
  float uv = bf2f(U[row * (size_t)K + src]);
  float v = gv * (1.f / (1.f + __expf(-gv))) * uv;
  float mx = v, mn = v;
  int lane = t & 31, wid = t >> 5;
  for (int off = 16; off; off >>= 1) {
    mx = fmaxf(mx, __shfl_xor(mx, off, 32));
    mn = fminf(mn, __shfl_xor(mn, off, 32));
  }
  __shared__ float smx[4], smn[4];
  if (!lane) { smx[wid] = mx; smn[wid] = mn; }
  __syncthreads();
  mx = fmaxf(fmaxf(smx[0], smx[1]), fmaxf(smx[2], smx[3]));
  mn = fminf(fminf(smn[0], smn[1]), fminf(smn[2], smn[3]));
  float out = v;
  if (g + 1 < (int)gridDim.y) {
    float scale = fmaxf(mx - mn, 1e-5f) * (1.f / 15.f);
    float inv = 1.f / scale;
    float base = fminf(fmaxf(rintf(-mn * inv), 0.f), 15.f);
    out = (fminf(fmaxf(rintf(v * inv) + base, 0.f), 15.f) - base) * scale;
  }
  Dq[row * (size_t)K + colp] = f2bf(out);
}

// ---------------------------------------------------------------------------
// NT GEMM: C[M,N] = A[M,K] * B[N,K]^T  (bf16 in, f32 accumulate via WMMA)
// 128x256 block tile, BK=32, 8 waves (2x4), each wave 64x64 (16 C tiles):
// 16 WMMA per 16 ds_load_b128 per K-step.  Double-buffered LDS fed with
// async global->LDS copies (6 per thread per stage).
// ---------------------------------------------------------------------------
#define BM 128
#define BN 256
#define BK 32

template <bool WRITE_BF16>
__global__ __launch_bounds__(256) void gemm_nt_bf16(
    const bf16_t* __restrict__ A, const bf16_t* __restrict__ Bw,
    const float* __restrict__ Res, void* __restrict__ C, int M, int N, int K) {
  __shared__ __align__(16) bf16_t As[2][BM * BK];   // 16 KB
  __shared__ __align__(16) bf16_t Bs[2][BN * BK];   // 32 KB

  const int t = threadIdx.x;
  const int wid = t >> 5, lane = t & 31;
  const int lh = lane >> 4, ln = lane & 15;
  const int wm = (wid >> 2) * 64;     // wave M offset in tile (0/64)
  const int wn = (wid & 3) * 64;      // wave N offset in tile (0..192)
  const size_t bm = (size_t)blockIdx.y * BM;
  const size_t bn = (size_t)blockIdx.x * BN;

  f32x8 acc[4][4];
#pragma unroll
  for (int i = 0; i < 4; ++i)
#pragma unroll
    for (int j = 0; j < 4; ++j)
#pragma unroll
      for (int r = 0; r < 8; ++r) acc[i][j][r] = 0.f;

  auto stage_load = [&](int s, int k0) {
#pragma unroll
    for (int i = 0; i < 2; ++i) {    // A: 128x32 = 512 x 16B chunks
      int lin = t + i * 256;
      int row = lin >> 2;
      int col = (lin & 3) * 8;
      copy16_g2l(&As[s][row * BK + col], A + (bm + row) * K + k0 + col);
    }
#pragma unroll
    for (int i = 0; i < 4; ++i) {    // B: 256x32 = 1024 x 16B chunks
      int lin = t + i * 256;
      int row = lin >> 2;
      int col = (lin & 3) * 8;
      copy16_g2l(&Bs[s][row * BK + col], Bw + (bn + row) * K + k0 + col);
    }
  };

  const int nk = K >> 5;
  stage_load(0, 0);
  for (int kt = 0; kt < nk; ++kt) {
    int cur = kt & 1;
    if (kt + 1 < nk) {
      stage_load(cur ^ 1, (kt + 1) << 5);
#if USE_ASYNC
      __builtin_amdgcn_s_wait_asynccnt(6);   // keep next stage in flight
#endif
    } else {
#if USE_ASYNC
      __builtin_amdgcn_s_wait_asynccnt(0);
#endif
    }
    __syncthreads();

    bf16x16 af[4], bfr[4];
#pragma unroll
    for (int mt = 0; mt < 4; ++mt)
      af[mt] = load_frag(&As[cur][(wm + mt * 16 + ln) * BK], lh);
#pragma unroll
    for (int nt = 0; nt < 4; ++nt)
      bfr[nt] = load_frag(&Bs[cur][(wn + nt * 16 + ln) * BK], lh);
#pragma unroll
    for (int mt = 0; mt < 4; ++mt)
#pragma unroll
      for (int nt = 0; nt < 4; ++nt)
        acc[mt][nt] = wmma_bf16(af[mt], bfr[nt], acc[mt][nt]);

    __syncthreads();
  }

  // Epilogue: C layout — VGPR r: lanes 0-15 -> M=r, lanes 16-31 -> M=r+8
#pragma unroll
  for (int mt = 0; mt < 4; ++mt)
#pragma unroll
    for (int nt = 0; nt < 4; ++nt)
#pragma unroll
      for (int r = 0; r < 8; ++r) {
        size_t row = bm + wm + mt * 16 + r + lh * 8;
        size_t col = bn + wn + nt * 16 + ln;
        size_t off = row * (size_t)N + col;
        float v = acc[mt][nt][r];
        if (Res) v += Res[off];
        if (WRITE_BF16) ((bf16_t*)C)[off] = f2bf(v);
        else            ((float*)C)[off] = v;
      }
}

// ---------------------------------------------------------------------------
// In-place RoPE on Q and K (bf16, rows = (b*S+s), 16 heads x 128 dims)
// ---------------------------------------------------------------------------
__global__ __launch_bounds__(256) void rope_inplace(bf16_t* __restrict__ Q,
                                                    bf16_t* __restrict__ Kk,
                                                    int S) {
  int row = blockIdx.x;
  int s = row % S;
  bf16_t* q = Q + (size_t)row * 2048;
  bf16_t* k = Kk + (size_t)row * 2048;
#pragma unroll
  for (int i = 0; i < 4; ++i) {
    int p = threadIdx.x + i * 256;     // 1024 (head, pair) slots
    int hh = p >> 6, d = p & 63;
    float invf = __expf(-(float)d * (9.210340371976184f / 64.f)); // 10000^(-d/64)
    float ang = (float)s * invf;
    float cs = __cosf(ang), sn = __sinf(ang);
    int b0 = hh * 128 + d;
    float q1 = bf2f(q[b0]), q2 = bf2f(q[b0 + 64]);
    q[b0]      = f2bf(q1 * cs - q2 * sn);
    q[b0 + 64] = f2bf(q2 * cs + q1 * sn);
    float k1 = bf2f(k[b0]), k2 = bf2f(k[b0 + 64]);
    k[b0]      = f2bf(k1 * cs - k2 * sn);
    k[b0 + 64] = f2bf(k2 * cs + k1 * sn);
  }
}

// ---------------------------------------------------------------------------
// Flash attention (causal), bf16 WMMA.  Q/K/V in (b*S+s, h*128+d) layout.
// Block: 128 q rows, 8 waves x 16 rows.  32-key tiles streamed through LDS.
// Out written f32 to (b*S+s, h*128+d).
// ---------------------------------------------------------------------------
__global__ __launch_bounds__(256) void flash_attn(
    const bf16_t* __restrict__ Q, const bf16_t* __restrict__ Kk,
    const bf16_t* __restrict__ V, float* __restrict__ Out, int S) {
  __shared__ __align__(16) bf16_t Ks[32 * 128];   // [key][d]
  __shared__ __align__(16) bf16_t Vs[128 * 32];   // transposed: [d][key]
  __shared__ __align__(16) bf16_t Ps[8][16 * 32]; // per-wave P tile

  const int t = threadIdx.x;
  const int wid = t >> 5, lane = t & 31;
  const int lh = lane >> 4, ln = lane & 15;
  const int bh = blockIdx.y;
  const int b = bh >> 4, h = bh & 15;
  const int qblk = blockIdx.x * 128;

  const bf16_t* Qp = Q + ((size_t)b * S) * 2048 + h * 128;
  const bf16_t* Kp = Kk + ((size_t)b * S) * 2048 + h * 128;
  const bf16_t* Vp = V + ((size_t)b * S) * 2048 + h * 128;

  // preload this wave's Q fragments (16 rows x 128 dims -> 4 K-chunks)
  bf16x16 qf[4];
  {
    int qrow = qblk + wid * 16 + ln;
#pragma unroll
    for (int c = 0; c < 4; ++c)
      qf[c] = load_frag(Qp + (size_t)qrow * 2048 + c * 32, lh);
  }

  f32x8 o[8];
#pragma unroll
  for (int j = 0; j < 8; ++j)
#pragma unroll
    for (int r = 0; r < 8; ++r) o[j][r] = 0.f;
  float m_i[8], l_i[8];
#pragma unroll
  for (int r = 0; r < 8; ++r) { m_i[r] = -3.0e38f; l_i[r] = 0.f; }

  const float sc = 0.08838834764831845f;  // 1/sqrt(128)
  const int kend = qblk + 128;            // causal limit for this block

  for (int kb = 0; kb < kend; kb += 32) {
    __syncthreads();  // protect LDS reuse across iterations
    // K tile: 32x128 = 512 x 16B chunks, async
#pragma unroll
    for (int i = 0; i < 2; ++i) {
      int lin = t + i * 256;
      int row = lin >> 4;
      int col = (lin & 15) * 8;
      copy16_g2l(&Ks[row * 128 + col], Kp + (size_t)(kb + row) * 2048 + col);
    }
    // V tile: load and transpose into Vs[d][key]
    {
      int krow = t >> 3;
      int c0 = (t & 7) * 16;
      const bf16_t* src = Vp + (size_t)(kb + krow) * 2048 + c0;
      bf16x8 a = *(const bf16x8*)src;
      bf16x8 b2 = *(const bf16x8*)(src + 8);
#pragma unroll
      for (int i = 0; i < 8; ++i) {
        Vs[(c0 + i) * 32 + krow] = a[i];
        Vs[(c0 + 8 + i) * 32 + krow] = b2[i];
      }
    }
#if USE_ASYNC
    __builtin_amdgcn_s_wait_asynccnt(0);
#endif
    __syncthreads();

    // scores: 16x32 per wave = two 16x16 WMMA accs over 4 head-dim chunks
    f32x8 s0, s1;
#pragma unroll
    for (int r = 0; r < 8; ++r) { s0[r] = 0.f; s1[r] = 0.f; }
#pragma unroll
    for (int c = 0; c < 4; ++c) {
      bf16x16 k0 = load_frag(&Ks[ln * 128 + c * 32], lh);
      bf16x16 k1 = load_frag(&Ks[(16 + ln) * 128 + c * 32], lh);
      s0 = wmma_bf16(qf[c], k0, s0);
      s1 = wmma_bf16(qf[c], k1, s1);
    }

    // online softmax (row stats across 16 lanes of each half-wave)
    float alpha[8];
#pragma unroll
    for (int r = 0; r < 8; ++r) {
      int rowg = qblk + wid * 16 + r + lh * 8;
      float v0 = s0[r] * sc; if (kb + ln > rowg)      v0 = -1e9f;
      float v1 = s1[r] * sc; if (kb + 16 + ln > rowg) v1 = -1e9f;
      float mx = fmaxf(v0, v1);
      for (int off = 8; off; off >>= 1) mx = fmaxf(mx, __shfl_xor(mx, off, 16));
      float mnew = fmaxf(m_i[r], mx);
      float a = __expf(m_i[r] - mnew);
      float p0 = __expf(v0 - mnew);
      float p1 = __expf(v1 - mnew);
      float rs = p0 + p1;
      for (int off = 8; off; off >>= 1) rs += __shfl_xor(rs, off, 16);
      l_i[r] = l_i[r] * a + rs;
      m_i[r] = mnew;
      alpha[r] = a;
      Ps[wid][(r + lh * 8) * 32 + ln] = f2bf(p0);
      Ps[wid][(r + lh * 8) * 32 + 16 + ln] = f2bf(p1);
    }
#pragma unroll
    for (int j = 0; j < 8; ++j)
#pragma unroll
      for (int r = 0; r < 8; ++r) o[j][r] *= alpha[r];

    // P (16x32) @ V (32x128) -> accumulate into o
    bf16x16 pf = load_frag(&Ps[wid][ln * 32], lh);
#pragma unroll
    for (int j = 0; j < 8; ++j) {
      bf16x16 vf = load_frag(&Vs[(j * 16 + ln) * 32], lh);
      o[j] = wmma_bf16(pf, vf, o[j]);
    }
  }

  float invl[8];
#pragma unroll
  for (int r = 0; r < 8; ++r) invl[r] = 1.f / l_i[r];
#pragma unroll
  for (int j = 0; j < 8; ++j)
#pragma unroll
    for (int r = 0; r < 8; ++r) {
      int row = qblk + wid * 16 + r + lh * 8;
      int col = j * 16 + ln;
      Out[((size_t)b * S + row) * 2048 + h * 128 + col] = o[j][r] * invl[r];
    }
}

// ---------------------------------------------------------------------------
// Orchestration
// ---------------------------------------------------------------------------
extern "C" void kernel_launch(void* const* d_in, const int* in_sizes, int n_in,
                              void* d_out, int out_size, void* d_ws,
                              size_t ws_size, hipStream_t stream) {
  (void)in_sizes; (void)n_in; (void)out_size; (void)ws_size;
  constexpr int Bz = 2, S = 2048, Hh = 2048, DFF = 8192;
  const int M = Bz * S;  // 4096

  const float* hidden = (const float*)d_in[0];
  const float* ln1 = (const float*)d_in[1];
  const float* ln2 = (const float*)d_in[2];
  const float* wq = (const float*)d_in[3];
  const float* wk = (const float*)d_in[4];
  const float* wv = (const float*)d_in[5];
  const float* wo = (const float*)d_in[6];
  const float* wg = (const float*)d_in[7];
  const float* wu = (const float*)d_in[8];
  const float* wd = (const float*)d_in[9];
  const int* idx_ain = (const int*)d_in[10];
  const int* idx_aout = (const int*)d_in[11];
  const int* idx_min = (const int*)d_in[12];
  const int* idx_mdn = (const int*)d_in[13];

  uint8_t* p = (uint8_t*)d_ws;
  auto alloc = [&](size_t n) -> void* {
    void* r = p; p += (n + 255) & ~(size_t)255; return r;
  };
  bf16_t* wqB = (bf16_t*)alloc((size_t)Hh * Hh * 2);
  bf16_t* wkB = (bf16_t*)alloc((size_t)Hh * Hh * 2);
  bf16_t* wvB = (bf16_t*)alloc((size_t)Hh * Hh * 2);
  bf16_t* woB = (bf16_t*)alloc((size_t)Hh * Hh * 2);
  bf16_t* wgB = (bf16_t*)alloc((size_t)DFF * Hh * 2);
  bf16_t* wuB = (bf16_t*)alloc((size_t)DFF * Hh * 2);
  bf16_t* wdB = (bf16_t*)alloc((size_t)Hh * DFF * 2);
  float* norm = (float*)alloc((size_t)M * Hh * 4);   // reused: attn out, norm2
  bf16_t* xq = (bf16_t*)alloc((size_t)M * Hh * 2);   // reused for all H-wide acts
  bf16_t* qb = (bf16_t*)alloc((size_t)M * Hh * 2);
  bf16_t* kb = (bf16_t*)alloc((size_t)M * Hh * 2);
  bf16_t* vb = (bf16_t*)alloc((size_t)M * Hh * 2);
  float* h1 = (float*)alloc((size_t)M * Hh * 4);
  bf16_t* gB = (bf16_t*)alloc((size_t)M * DFF * 2);
  bf16_t* uB = (bf16_t*)alloc((size_t)M * DFF * 2);
  bf16_t* dq = (bf16_t*)alloc((size_t)M * DFF * 2);

  dim3 bq(128);
  // 1. weight reorder+quant -> bf16
  reorder_quant_k<<<dim3(Hh, Hh / 128), bq, 0, stream>>>(wq, idx_ain, wqB, Hh, 1);
  reorder_quant_k<<<dim3(Hh, Hh / 128), bq, 0, stream>>>(wk, idx_ain, wkB, Hh, 1);
  reorder_quant_k<<<dim3(Hh, Hh / 128), bq, 0, stream>>>(wv, idx_ain, wvB, Hh, 1);
  reorder_quant_k<<<dim3(Hh, Hh / 128), bq, 0, stream>>>(wo, idx_aout, woB, Hh, 1);
  reorder_quant_k<<<dim3(DFF, Hh / 128), bq, 0, stream>>>(wg, idx_min, wgB, Hh, 1);
  reorder_quant_k<<<dim3(DFF, Hh / 128), bq, 0, stream>>>(wu, idx_min, wuB, Hh, 1);
  reorder_quant_k<<<dim3(Hh, DFF / 128), bq, 0, stream>>>(wd, idx_mdn, wdB, DFF, 1);

  // 2-3. rmsnorm + activation reorder/quant
  rmsnorm_k<<<M, 256, 0, stream>>>(hidden, ln1, norm);
  reorder_quant_k<<<dim3(M, Hh / 128), bq, 0, stream>>>(norm, idx_ain, xq, Hh, 0);

  // 4. QKV projections (WMMA)
  dim3 gb(256), gg(Hh / BN, M / BM);
  gemm_nt_bf16<true><<<gg, gb, 0, stream>>>(xq, wqB, nullptr, qb, M, Hh, Hh);
  gemm_nt_bf16<true><<<gg, gb, 0, stream>>>(xq, wkB, nullptr, kb, M, Hh, Hh);
  gemm_nt_bf16<true><<<gg, gb, 0, stream>>>(xq, wvB, nullptr, vb, M, Hh, Hh);

  // 5. RoPE (in place on Q,K)
  rope_inplace<<<M, 256, 0, stream>>>(qb, kb, S);

  // 6. causal flash attention -> f32 (reuses `norm` buffer)
  flash_attn<<<dim3(S / 128, Bz * 16), 256, 0, stream>>>(qb, kb, vb, norm, S);

  // 7-8. output projection + residual
  reorder_quant_k<<<dim3(M, Hh / 128), bq, 0, stream>>>(norm, idx_aout, xq, Hh, 0);
  gemm_nt_bf16<false><<<gg, gb, 0, stream>>>(xq, woB, hidden, h1, M, Hh, Hh);

  // 9-11. MLP
  rmsnorm_k<<<M, 256, 0, stream>>>(h1, ln2, norm);
  reorder_quant_k<<<dim3(M, Hh / 128), bq, 0, stream>>>(norm, idx_min, xq, Hh, 0);
  dim3 gg2(DFF / BN, M / BM);
  gemm_nt_bf16<true><<<gg2, gb, 0, stream>>>(xq, wgB, nullptr, gB, M, DFF, Hh);
  gemm_nt_bf16<true><<<gg2, gb, 0, stream>>>(xq, wuB, nullptr, uB, M, DFF, Hh);

  // 12. SiLU*up -> reorder/quant
  silu_mul_quant_k<<<dim3(M, DFF / 128), bq, 0, stream>>>(gB, uB, idx_mdn, dq, DFF);

  // 13. down projection + residual -> output
  gemm_nt_bf16<false><<<gg, gb, 0, stream>>>(dq, wdB, h1, (float*)d_out, M, Hh, DFF);
}
